// MoELayer_11871289606928
// MI455X (gfx1250) — matrix operand order repaired and checked by reading, not compiled, not run on hardware
//
#include <hip/hip_runtime.h>
#include <hip/hip_bf16.h>

typedef __attribute__((ext_vector_type(16))) _Float16 v16h;
typedef __attribute__((ext_vector_type(8)))  float    v8f;

#define NTOK   16384
#define DMODEL 1024
#define DFF    4096
#define NEXP   8
#define CAP    320
#define SINK   (NEXP * CAP)

#define BM 64
#define BN 128
#define BK 32
#define LDA 40   // A LDS row stride in halves (80 B, 16B aligned)
#define LDB 20   // B LDS col stride in u32 K-pairs (80 B, 16B aligned)

#define WMMA_F16(a, b, c) \
    __builtin_amdgcn_wmma_f32_16x16x32_f16(false, (a), false, (b), (short)0, (c), false, false)

// ---------------------------------------------------------------------------
// Router: one wave32 per token, fp32 (argmax ties decide routing -> keep f32).
// ---------------------------------------------------------------------------
__global__ void moe_router(const float* __restrict__ x,
                           const float* __restrict__ rw,
                           const float* __restrict__ rb,
                           int* __restrict__ eidx,
                           float* __restrict__ probsSum) {
    const int wave = threadIdx.x >> 5;
    const int lane = threadIdx.x & 31;
    const int t    = blockIdx.x * 8 + wave;

    float acc[8] = {0.f, 0.f, 0.f, 0.f, 0.f, 0.f, 0.f, 0.f};
    const float* xr = x + (size_t)t * DMODEL;
    for (int k = lane; k < DMODEL; k += 32) {
        float xv = xr[k];
        const float4* w4 = (const float4*)(rw + (size_t)k * NEXP);
        float4 w0 = w4[0], w1v = w4[1];
        acc[0] += xv * w0.x;  acc[1] += xv * w0.y;
        acc[2] += xv * w0.z;  acc[3] += xv * w0.w;
        acc[4] += xv * w1v.x; acc[5] += xv * w1v.y;
        acc[6] += xv * w1v.z; acc[7] += xv * w1v.w;
    }
#pragma unroll
    for (int e = 0; e < 8; ++e)
#pragma unroll
        for (int off = 16; off > 0; off >>= 1)
            acc[e] += __shfl_down(acc[e], off, 32);

    __shared__ float sprob[8][8];
    if (lane == 0) {
        float logit[8], mx = -1e30f;
#pragma unroll
        for (int e = 0; e < 8; ++e) { logit[e] = acc[e] + rb[e]; mx = fmaxf(mx, logit[e]); }
        float p[8], sum = 0.f;
#pragma unroll
        for (int e = 0; e < 8; ++e) { p[e] = expf(logit[e] - mx); sum += p[e]; }
        float inv = 1.f / sum;
        int arg = 0; float best = logit[0];
#pragma unroll
        for (int e = 1; e < 8; ++e) if (logit[e] > best) { best = logit[e]; arg = e; }
        eidx[t] = arg;
#pragma unroll
        for (int e = 0; e < 8; ++e) sprob[wave][e] = p[e] * inv;
    }
    __syncthreads();
    if (threadIdx.x < 8) {
        float s = 0.f;
#pragma unroll
        for (int w = 0; w < 8; ++w) s += sprob[w][threadIdx.x];
        atomicAdd(&probsSum[threadIdx.x], s);
    }
}

// ---------------------------------------------------------------------------
// Pass A: per-block (1024 tokens) per-expert counts via wave32 ballots.
// ---------------------------------------------------------------------------
__global__ void moe_count(const int* __restrict__ eidx, int* __restrict__ blockCounts) {
    const int lane = threadIdx.x;          // 32 threads
    const int base = blockIdx.x * 1024;
    int cnt[8] = {0, 0, 0, 0, 0, 0, 0, 0};
    for (int g = 0; g < 32; ++g) {
        int e = eidx[base + g * 32 + lane];
#pragma unroll
        for (int q = 0; q < 8; ++q) {
            unsigned m = __builtin_amdgcn_ballot_w32(e == q);
            cnt[q] += __popc(m);
        }
    }
    if (lane < 8) blockCounts[blockIdx.x * 8 + lane] = cnt[lane];
}

// ---------------------------------------------------------------------------
// Pass B: exclusive prefix over the 16 block counts per expert + totals.
// ---------------------------------------------------------------------------
__global__ void moe_scan(const int* __restrict__ blockCounts,
                         int* __restrict__ blockOffsets,
                         int* __restrict__ totals) {
    int e = threadIdx.x;
    if (e < 8) {
        int off = 0;
        for (int b = 0; b < 16; ++b) {
            blockOffsets[b * 8 + e] = off;
            off += blockCounts[b * 8 + e];
        }
        totals[e] = off;
    }
}

// ---------------------------------------------------------------------------
// Pass C: exact token-order position within expert (ballot + lt-popcount).
// ---------------------------------------------------------------------------
__global__ void moe_assign(const int* __restrict__ eidx,
                           const int* __restrict__ blockOffsets,
                           int* __restrict__ slotToToken) {
    const int lane = threadIdx.x;
    const int blk  = blockIdx.x;
    const int base = blk * 1024;
    int cnt[8];
#pragma unroll
    for (int q = 0; q < 8; ++q) cnt[q] = blockOffsets[blk * 8 + q];
    const unsigned ltmask = (1u << lane) - 1u;
    for (int g = 0; g < 32; ++g) {
        int t = base + g * 32 + lane;
        int e = eidx[t];
        unsigned mymask = 0;
        int mybase = 0;
        int newcnt[8];
#pragma unroll
        for (int q = 0; q < 8; ++q) {
            unsigned m = __builtin_amdgcn_ballot_w32(e == q);
            if (e == q) { mymask = m; mybase = cnt[q]; }
            newcnt[q] = cnt[q] + __popc(m);
        }
        int pos = mybase + __popc(mymask & ltmask);
        if (pos < CAP) slotToToken[e * CAP + pos] = t;
#pragma unroll
        for (int q = 0; q < 8; ++q) cnt[q] = newcnt[q];
    }
}

// ---------------------------------------------------------------------------
// Fragment loaders, CDNA5 WMMA layout (wave32), all 128-bit LDS loads.
//   A 16x32 f16 (row-major LDS): lane<16 -> M=lane,   K {0..7,16..23}
//                                lane>=16-> M=lane-16,K {8..15,24..31}
//   B 32x16 f16 (K-pair-packed, N-major LDS): lane<16 -> N=lane,    K {0..7,16..23}
//                                             lane>=16-> N=lane-16, K {8..15,24..31}
// ---------------------------------------------------------------------------
__device__ __forceinline__ v16h load_frag_a(const _Float16* __restrict__ As,
                                            int lane, int mOff) {
    int row = mOff + (lane & 15);
    int kb  = (lane < 16) ? 0 : 8;
    const _Float16* r = As + row * LDA + kb;
    union { v16h v; float4 f[2]; } u;
    u.f[0] = *(const float4*)r;          // K kb..kb+7
    u.f[1] = *(const float4*)(r + 16);   // K kb+16..kb+23
    return u.v;
}

__device__ __forceinline__ v16h load_frag_b(const unsigned* __restrict__ Bs,
                                            int lane, int nOff) {
    int col = nOff + (lane & 15);
    int kp  = (lane < 16) ? 0 : 4;       // u32 K-pair base
    const unsigned* c = Bs + col * LDB + kp;
    union { v16h v; float4 f[2]; } u;
    u.f[0] = *(const float4*)c;          // K 2kp..2kp+7
    u.f[1] = *(const float4*)(c + 8);    // K 2kp+16..2kp+23
    return u.v;
}

// ---------------------------------------------------------------------------
// FFN GEMM 1: h[e] = relu( gather(x)[320x1024] @ w1[e][1024x4096] + b1[e] )
// Double-buffered LDS (static indices via 2x-unrolled pipeline), all B
// fragments preloaded so the 4 WMMAs issue back-to-back.
// ---------------------------------------------------------------------------
__global__ void __launch_bounds__(256)
moe_ffn1(const float* __restrict__ x, const float* __restrict__ w1,
         const float* __restrict__ b1, const int* __restrict__ slotToToken,
         _Float16* __restrict__ h) {
    const int e     = blockIdx.z;
    const int mBase = blockIdx.y * BM;
    const int nBase = blockIdx.x * BN;
    __shared__ alignas(16) _Float16 As[2][BM * LDA];
    __shared__ alignas(16) unsigned Bs[2][BN * LDB];

    const int tid = threadIdx.x, lane = tid & 31, wid = tid >> 5;
    const int wm = wid >> 1, wn = wid & 1;        // 4 M-tiles x 2 N-halves
    v8f acc[4] = {{}, {}, {}, {}};

    const float* w1e = w1 + (size_t)e * DMODEL * DFF;
    const int ar = tid >> 2,  ac = (tid & 3) * 8;     // A: 8 f32 / thread
    const int bkp = tid >> 4, bc = (tid & 15) * 8;    // B: K-pair + 8 cols
    const int atok = slotToToken[e * CAP + mBase + ar];
    const float* aSrc = (atok >= 0) ? (x + (size_t)atok * DMODEL + ac) : nullptr;
    const float* bSrc = w1e + (size_t)(2 * bkp) * DFF + nBase + bc;

    // empty rows: zero once in both stages (single LDS writer per slot)
    if (!aSrc) {
        float4 z = {0.f, 0.f, 0.f, 0.f};
        *(float4*)&As[0][ar * LDA + ac] = z;
        *(float4*)&As[1][ar * LDA + ac] = z;
    }

    union { float4 f4[2]; float f[8]; } uA, uLo, uHi;

    auto loadRegs = [&](int k0) {
        if (aSrc) {
            uA.f4[0] = *(const float4*)(aSrc + k0);
            uA.f4[1] = *(const float4*)(aSrc + k0 + 4);
        }
        const float* bp = bSrc + (size_t)k0 * DFF;
        __builtin_prefetch(bp + (size_t)BK * DFF, 0, 1);
        uLo.f4[0] = *(const float4*)bp;
        uLo.f4[1] = *(const float4*)(bp + 4);
        uHi.f4[0] = *(const float4*)(bp + DFF);
        uHi.f4[1] = *(const float4*)(bp + DFF + 4);
    };
    auto storeStage = [&](int s) {
        if (aSrc) {
            union { float4 f4; _Float16 hh[8]; } pa;
#pragma unroll
            for (int i = 0; i < 8; ++i) pa.hh[i] = (_Float16)uA.f[i];
            *(float4*)&As[s][ar * LDA + ac] = pa.f4;    // one ds_store_b128
        }
#pragma unroll
        for (int i = 0; i < 8; ++i) {                   // 8x ds_store_b32
            union { _Float16 hh[2]; unsigned u; } p;
            p.hh[0] = (_Float16)uLo.f[i];               // K = 2*bkp
            p.hh[1] = (_Float16)uHi.f[i];               // K = 2*bkp+1
            Bs[s][(bc + i) * LDB + bkp] = p.u;
        }
    };
    auto compute = [&](int s) {
        // preload all fragments (10x ds_load_b128), then 4 WMMAs back-to-back
        v16h a  = load_frag_a(&As[s][0], lane, wm * 16);
        v16h b0 = load_frag_b(&Bs[s][0], lane, wn * 64);
        v16h b1f = load_frag_b(&Bs[s][0], lane, wn * 64 + 16);
        v16h b2f = load_frag_b(&Bs[s][0], lane, wn * 64 + 32);
        v16h b3f = load_frag_b(&Bs[s][0], lane, wn * 64 + 48);
        acc[0] = WMMA_F16(a, b0, acc[0]);
        acc[1] = WMMA_F16(a, b1f, acc[1]);
        acc[2] = WMMA_F16(a, b2f, acc[2]);
        acc[3] = WMMA_F16(a, b3f, acc[3]);
    };

    constexpr int ITERS = DMODEL / BK;                  // 32 (even)
    loadRegs(0);
    storeStage(0);
    for (int it = 0; it < ITERS; it += 2) {             // static buffer indices
        __syncthreads();
        loadRegs((it + 1) * BK);
        compute(0);
        storeStage(1);
        __syncthreads();
        const bool more = (it + 2 < ITERS);
        if (more) loadRegs((it + 2) * BK);
        compute(1);
        if (more) storeStage(0);
    }

    const float* b1e = b1 + e * DFF;
    const int rowAdd = (lane < 16) ? 0 : 8;
    _Float16* he = h + (size_t)e * CAP * DFF;
    int   nIdx[4];
    float bias[4];
#pragma unroll
    for (int j = 0; j < 4; ++j) {
        nIdx[j] = nBase + wn * 64 + j * 16 + (lane & 15);
        bias[j] = b1e[nIdx[j]];
    }
#pragma unroll
    for (int i = 0; i < 8; ++i) {
        int m = mBase + wm * 16 + i + rowAdd;
        _Float16* hrow = he + (size_t)m * DFF;
#pragma unroll
        for (int j = 0; j < 4; ++j)
            hrow[nIdx[j]] = (_Float16)fmaxf(acc[j][i] + bias[j], 0.f);
    }
}

// ---------------------------------------------------------------------------
// FFN GEMM 2: y = h[e][320x4096] @ w2[e][4096x1024] + b2[e]; scatter to out.
// ---------------------------------------------------------------------------
__global__ void __launch_bounds__(256)
moe_ffn2(const _Float16* __restrict__ h, const float* __restrict__ w2,
         const float* __restrict__ b2, const int* __restrict__ slotToToken,
         float* __restrict__ out) {
    const int e     = blockIdx.z;
    const int mBase = blockIdx.y * BM;
    const int nBase = blockIdx.x * BN;
    __shared__ alignas(16) _Float16 As[2][BM * LDA];
    __shared__ alignas(16) unsigned Bs[2][BN * LDB];

    const int tid = threadIdx.x, lane = tid & 31, wid = tid >> 5;
    const int wm = wid >> 1, wn = wid & 1;
    v8f acc[4] = {{}, {}, {}, {}};

    const _Float16* he = h + (size_t)e * CAP * DFF;
    const float* w2e = w2 + (size_t)e * DFF * DMODEL;
    const int ar = tid >> 2,  ac = (tid & 3) * 8;
    const int bkp = tid >> 4, bc = (tid & 15) * 8;
    const _Float16* aSrc = he + (size_t)(mBase + ar) * DFF + ac;
    const float*    bSrc = w2e + (size_t)(2 * bkp) * DMODEL + nBase + bc;

    float4 aR;                                    // 8 halves, already f16
    union { float4 f4[2]; float f[8]; } uLo, uHi;

    auto loadRegs = [&](int k0) {
        aR = *(const float4*)(aSrc + k0);
        const float* bp = bSrc + (size_t)k0 * DMODEL;
        __builtin_prefetch(bp + (size_t)BK * DMODEL, 0, 1);
        uLo.f4[0] = *(const float4*)bp;
        uLo.f4[1] = *(const float4*)(bp + 4);
        uHi.f4[0] = *(const float4*)(bp + DMODEL);
        uHi.f4[1] = *(const float4*)(bp + DMODEL + 4);
    };
    auto storeStage = [&](int s) {
        *(float4*)&As[s][ar * LDA + ac] = aR;     // one ds_store_b128
#pragma unroll
        for (int i = 0; i < 8; ++i) {
            union { _Float16 hh[2]; unsigned u; } p;
            p.hh[0] = (_Float16)uLo.f[i];
            p.hh[1] = (_Float16)uHi.f[i];
            Bs[s][(bc + i) * LDB + bkp] = p.u;
        }
    };
    auto compute = [&](int s) {
        v16h a  = load_frag_a(&As[s][0], lane, wm * 16);
        v16h b0 = load_frag_b(&Bs[s][0], lane, wn * 64);
        v16h b1f = load_frag_b(&Bs[s][0], lane, wn * 64 + 16);
        v16h b2f = load_frag_b(&Bs[s][0], lane, wn * 64 + 32);
        v16h b3f = load_frag_b(&Bs[s][0], lane, wn * 64 + 48);
        acc[0] = WMMA_F16(a, b0, acc[0]);
        acc[1] = WMMA_F16(a, b1f, acc[1]);
        acc[2] = WMMA_F16(a, b2f, acc[2]);
        acc[3] = WMMA_F16(a, b3f, acc[3]);
    };

    constexpr int ITERS = DFF / BK;                     // 128 (even)
    loadRegs(0);
    storeStage(0);
    for (int it = 0; it < ITERS; it += 2) {
        __syncthreads();
        loadRegs((it + 1) * BK);
        compute(0);
        storeStage(1);
        __syncthreads();
        const bool more = (it + 2 < ITERS);
        if (more) loadRegs((it + 2) * BK);
        compute(1);
        if (more) storeStage(0);
    }

    const float* b2e = b2 + e * DMODEL;
    const int rowAdd = (lane < 16) ? 0 : 8;
    int   nIdx[4];
    float bias[4];
#pragma unroll
    for (int j = 0; j < 4; ++j) {
        nIdx[j] = nBase + wn * 64 + j * 16 + (lane & 15);
        bias[j] = b2e[nIdx[j]];
    }
#pragma unroll
    for (int i = 0; i < 8; ++i) {
        int r = mBase + wm * 16 + i + rowAdd;
        int tok = slotToToken[e * CAP + r];
        if (tok >= 0) {
            float* orow = out + (size_t)tok * DMODEL;
#pragma unroll
            for (int j = 0; j < 4; ++j)
                orow[nIdx[j]] = acc[j][i] + bias[j];
        }
    }
}

// ---------------------------------------------------------------------------
// Load-balance loss: E * sum_e (count_e/T)*(probsSum_e/T)
// ---------------------------------------------------------------------------
__global__ void moe_loss(const int* __restrict__ totals,
                         const float* __restrict__ probsSum,
                         float* __restrict__ out_loss) {
    if (threadIdx.x == 0) {
        float s = 0.f;
        for (int e = 0; e < 8; ++e)
            s += ((float)totals[e] / (float)NTOK) * (probsSum[e] / (float)NTOK);
        *out_loss = (float)NEXP * s;
    }
}

// ---------------------------------------------------------------------------
extern "C" void kernel_launch(void* const* d_in, const int* in_sizes, int n_in,
                              void* d_out, int out_size, void* d_ws, size_t ws_size,
                              hipStream_t stream) {
    const float* x  = (const float*)d_in[0];
    const float* rw = (const float*)d_in[1];
    const float* rb = (const float*)d_in[2];
    const float* w1 = (const float*)d_in[3];
    const float* b1 = (const float*)d_in[4];
    const float* w2 = (const float*)d_in[5];
    const float* b2 = (const float*)d_in[6];
    float* out = (float*)d_out;

    // workspace layout
    char* p = (char*)d_ws;
    int*   eidx         = (int*)p;   p += (size_t)NTOK * 4;
    int*   blockCounts  = (int*)p;   p += 16 * 8 * 4;
    int*   blockOffsets = (int*)p;   p += 16 * 8 * 4;
    int*   totals       = (int*)p;   p += 8 * 4;
    float* probsSum     = (float*)p; p += 8 * 4;
    int*   slotToToken  = (int*)p;   p += SINK * 4;
    p = (char*)(((uintptr_t)p + 255) & ~(uintptr_t)255);
    _Float16* hbuf = (_Float16*)p;   // 8*320*4096 halves = 21 MB

    hipMemsetAsync(d_out, 0, (size_t)out_size * sizeof(float), stream);
    hipMemsetAsync(probsSum, 0, 8 * sizeof(float), stream);
    hipMemsetAsync(slotToToken, 0xFF, SINK * sizeof(int), stream);  // -1

    moe_router<<<NTOK / 8, 256, 0, stream>>>(x, rw, rb, eidx, probsSum);
    moe_count <<<16, 32, 0, stream>>>(eidx, blockCounts);
    moe_scan  <<<1, 8, 0, stream>>>(blockCounts, blockOffsets, totals);
    moe_assign<<<16, 32, 0, stream>>>(eidx, blockOffsets, slotToToken);

    moe_ffn1<<<dim3(DFF / BN, CAP / BM, NEXP), 256, 0, stream>>>(x, w1, b1, slotToToken, hbuf);
    moe_ffn2<<<dim3(DMODEL / BN, CAP / BM, NEXP), 256, 0, stream>>>(hbuf, w2, b2, slotToToken, out);

    moe_loss<<<1, 32, 0, stream>>>(totals, probsSum, out + (size_t)NTOK * DMODEL);
}
